// MultiHeadAttention_87239375716860
// MI455X (gfx1250) — compile-verified, matrix operand order checked by reading, not run on hardware
//
#include <hip/hip_runtime.h>

typedef __attribute__((ext_vector_type(16))) __bf16 v16bf;
typedef __attribute__((ext_vector_type(8)))  float  v8f;
typedef __attribute__((ext_vector_type(4)))  unsigned int u32x4;
typedef __attribute__((ext_vector_type(8)))  int  i32x8;
typedef __attribute__((ext_vector_type(4)))  int  i32x4;
typedef __bf16 bf16_t;

#define DMODEL 1024
#define SB     8192      // SEQ*BATCH rows
#define SEQL   2048
#define BATCHN 4
#define NH     16
#define DKH    64

#if defined(__has_builtin)
#  if __has_builtin(__builtin_amdgcn_tensor_load_to_lds) && \
      __has_builtin(__builtin_amdgcn_s_wait_tensorcnt)
#    define HAVE_TDM 1
#  endif
#endif

// ---------------------------------------------------------------------------
// WMMA helper: D = A(16x32 bf16) * B(32x16 bf16) + C(16x16 f32)
// ---------------------------------------------------------------------------
static __device__ __forceinline__ v8f wmma_bf16(v16bf a, v16bf b, v8f c) {
  return __builtin_amdgcn_wmma_f32_16x16x32_bf16(false, a, false, b, (short)0, c,
                                                 false, false);
}

// Per-lane A/B fragment (16 bf16) from an f32 row. Lane layout (ISA 7.12.2):
// lanes 0-15 take K {0..7,16..23} (o=0), lanes 16-31 {8..15,24..31} (o=8).
static __device__ __forceinline__ v16bf load_frag_f32(const float* __restrict__ p, int o) {
  const float4* a = (const float4*)(p + o);
  const float4* b = (const float4*)(p + o + 16);
  float4 f0 = a[0], f1 = a[1], f2 = b[0], f3 = b[1];
  v16bf r;
  r[0]  = (__bf16)f0.x; r[1]  = (__bf16)f0.y; r[2]  = (__bf16)f0.z; r[3]  = (__bf16)f0.w;
  r[4]  = (__bf16)f1.x; r[5]  = (__bf16)f1.y; r[6]  = (__bf16)f1.z; r[7]  = (__bf16)f1.w;
  r[8]  = (__bf16)f2.x; r[9]  = (__bf16)f2.y; r[10] = (__bf16)f2.z; r[11] = (__bf16)f2.w;
  r[12] = (__bf16)f3.x; r[13] = (__bf16)f3.y; r[14] = (__bf16)f3.z; r[15] = (__bf16)f3.w;
  return r;
}

// Same fragment from a bf16 row (global or LDS): two 16-byte loads.
static __device__ __forceinline__ v16bf load_frag_bf16(const bf16_t* p, int o) {
  union { uint4 u[2]; v16bf v; } x;
  x.u[0] = *(const uint4*)(p + o);
  x.u[1] = *(const uint4*)(p + o + 16);
  return x.v;
}

// ---------------------------------------------------------------------------
// GEMM:  Y[m,n] = sum_d X[m,d] * W[n,d] + bias[n]      (X: SBxDM, W: DMxDM)
// One wave per 32x32 tile (2x2 WMMA subtiles -> 4 WMMA per 4 fragment loads,
// ~8 flop/byte of operand traffic). mode: 0 = bf16 row-major out,
// 1 = bf16 transposed out Vt[b][e][s] (m = s*4+b), 2 = f32 row-major out.
// ---------------------------------------------------------------------------
__global__ __launch_bounds__(256) void gemm_kernel(const float* __restrict__ X,
                                                   const float* __restrict__ Wm,
                                                   const float* __restrict__ bias,
                                                   bf16_t* __restrict__ outRow,
                                                   bf16_t* __restrict__ outTr,
                                                   float* __restrict__ outF32,
                                                   int mode) {
  const int lane = threadIdx.x & 31;
  const int wid  = blockIdx.x * 8 + (threadIdx.x >> 5);   // 0..8191
  const int tM   = wid >> 5;                              // 0..255
  const int tN   = wid & 31;                              // 0..31
  const int m0   = tM << 5;
  const int n0   = tN << 5;
  const int l15  = lane & 15;
  const int hi   = lane >> 4;
  const int o    = hi ? 8 : 0;

  const float* ar0 = X  + (size_t)(m0 + l15) * DMODEL;
  const float* ar1 = X  + (size_t)(m0 + 16 + l15) * DMODEL;
  const float* br0 = Wm + (size_t)(n0 + l15) * DMODEL;
  const float* br1 = Wm + (size_t)(n0 + 16 + l15) * DMODEL;

  v8f acc[2][2];
#pragma unroll
  for (int i = 0; i < 2; ++i)
#pragma unroll
    for (int j = 0; j < 2; ++j) acc[i][j] = (v8f){};

  for (int kk = 0; kk < DMODEL; kk += 32) {
    v16bf a0 = load_frag_f32(ar0 + kk, o);
    v16bf a1 = load_frag_f32(ar1 + kk, o);
    v16bf b0 = load_frag_f32(br0 + kk, o);
    v16bf b1 = load_frag_f32(br1 + kk, o);
    acc[0][0] = wmma_bf16(a0, b0, acc[0][0]);
    acc[0][1] = wmma_bf16(a0, b1, acc[0][1]);
    acc[1][0] = wmma_bf16(a1, b0, acc[1][0]);
    acc[1][1] = wmma_bf16(a1, b1, acc[1][1]);
  }

#pragma unroll
  for (int ni = 0; ni < 2; ++ni) {
    const int   n  = n0 + ni * 16 + l15;
    const float bv = bias[n];
#pragma unroll
    for (int mi = 0; mi < 2; ++mi) {
      const int mb = m0 + mi * 16 + (hi << 3);
      if (mode == 0) {
#pragma unroll
        for (int v = 0; v < 8; ++v)
          outRow[(size_t)(mb + v) * DMODEL + n] = (__bf16)(acc[mi][ni][v] + bv);
      } else if (mode == 1) {
#pragma unroll
        for (int v = 0; v < 8; ++v) {
          const int m = mb + v;                 // m = s*BATCH + b
          outTr[((size_t)(m & 3) * DMODEL + n) * SEQL + (m >> 2)] =
              (__bf16)(acc[mi][ni][v] + bv);
        }
      } else {
#pragma unroll
        for (int v = 0; v < 8; ++v)
          outF32[(size_t)(mb + v) * DMODEL + n] = acc[mi][ni][v] + bv;
      }
    }
  }
}

// ---------------------------------------------------------------------------
// Fused attention with softmax over the HEADS axis (reference quirk).
// Block = 8 waves = one (b, 16-row q tile). Wave w owns heads {2w, 2w+1}.
// The 32x1024 K slab for each k-tile is shared by all 8 waves, so it is
// staged once into LDS by the Tensor Data Mover (wave 0 issues the 2D D#,
// completes with s_wait_tensorcnt before the publishing barrier).
// Per tile: scores (WMMA), exp, cross-head denominator via ds_add_f32,
// normalize, LDS-stage probs bf16 (C-layout -> A-layout), attn@V (WMMA).
// ---------------------------------------------------------------------------
__global__ __launch_bounds__(256) void attn_kernel(const bf16_t* __restrict__ Qb,
                                                   const bf16_t* __restrict__ Kb,
                                                   const bf16_t* __restrict__ Vt,
                                                   float* __restrict__ AttnOut) {
  __shared__ float denom[512];                        // [16 q][32 k]
  __shared__ __align__(16) bf16_t attnLds[NH * 512];  // [h][16 q][32 k]
  __shared__ __align__(16) bf16_t Klds[32 * DMODEL];  // [32 k][1024 e]  64 KB

  const int lane = threadIdx.x & 31;
  const int wave = threadIdx.x >> 5;
  const int b    = blockIdx.x & 3;
  const int q0   = (blockIdx.x >> 2) << 4;
  const int l15  = lane & 15;
  const int hi   = lane >> 4;
  const int o    = hi ? 8 : 0;
  const int h0   = wave << 1;
  const int qr   = hi << 3;                           // C-layout M base per lane

  // Q fragments are k-loop invariant: per head, dk=64 -> 2 K-chunks of 32.
  v16bf qf[2][2];
#pragma unroll
  for (int hh = 0; hh < 2; ++hh) {
    const bf16_t* qrow =
        Qb + (size_t)((q0 + l15) * BATCHN + b) * DMODEL + (h0 + hh) * DKH;
    qf[hh][0] = load_frag_bf16(qrow, o);
    qf[hh][1] = load_frag_bf16(qrow + 32, o);
  }

  v8f outacc[2][4];
#pragma unroll
  for (int hh = 0; hh < 2; ++hh)
#pragma unroll
    for (int nt = 0; nt < 4; ++nt) outacc[hh][nt] = (v8f){};

  for (int k0 = 0; k0 < SEQL; k0 += 32) {
    __syncthreads();   // previous Klds/denom consumers are done

#if defined(HAVE_TDM)
    if (wave == 0) {
      // D# group0: count=1, lds_addr, 57b global_addr, type=2 ("image")
      const unsigned long long ga =
          (unsigned long long)(const void*)(Kb + (size_t)(k0 * BATCHN + b) * DMODEL);
      const unsigned lds_addr = (unsigned)(size_t)(const void*)Klds;
      u32x4 g0;
      g0[0] = 1u;                                   // count=1
      g0[1] = lds_addr;
      g0[2] = (unsigned)ga;
      g0[3] = ((unsigned)(ga >> 32) & 0x1FFFFFFu) | (2u << 30);
      // D# group1: data_size=1 (2B), tensor_dim0=4096, tensor_dim1=8192,
      // tile_dim0=1024, tile_dim1=32, tensor_dim0_stride=4096 (batch interleave)
      i32x8 g1;
      g1[0] = (1 << 16);                            // data_size = 2 bytes
      g1[1] = (int)(4096u << 16);                   // tensor_dim0[15:0] << 16
      g1[2] = (int)((4096u >> 16) | (8192u << 16)); // dim0 hi | tensor_dim1 lo
      g1[3] = (int)((8192u >> 16) | (1024u << 16)); // dim1 hi | tile_dim0
      g1[4] = 32;                                   // tile_dim1 (tile_dim2=0)
      g1[5] = 4096;                                 // tensor_dim0_stride lo
      g1[6] = 0;                                    // stride hi | dim1_stride lo
      g1[7] = 0;
      i32x4 z4 = (i32x4){};
#if __clang_major__ >= 23
      i32x8 z8 = (i32x8){};
      __builtin_amdgcn_tensor_load_to_lds(g0, g1, z4, z4, z8, 0);
#else
      __builtin_amdgcn_tensor_load_to_lds(g0, g1, z4, z4, 0);
#endif
    }
#else
    // Fallback: cooperative copy of the K slab (32 rows x 1024 bf16).
    for (int i = threadIdx.x; i < 32 * DMODEL / 8; i += 256) {
      const int r = i >> 7, c = i & 127;
      ((uint4*)Klds)[i] =
          *(const uint4*)(Kb + (size_t)((k0 + r) * BATCHN + b) * DMODEL + c * 8);
    }
#endif

    for (int i = threadIdx.x; i < 512; i += 256) denom[i] = 0.0f;

#if defined(HAVE_TDM)
    if (wave == 0) __builtin_amdgcn_s_wait_tensorcnt(0);
#endif
    __syncthreads();   // publish Klds + zeroed denom

    // ---- scores + exp (K operands now from LDS) ----
    v8f e[2][2];
#pragma unroll
    for (int hh = 0; hh < 2; ++hh) {
      const int h = h0 + hh;
      const bf16_t* kr0 = &Klds[(size_t)l15 * DMODEL + h * DKH];
      const bf16_t* kr1 = &Klds[(size_t)(16 + l15) * DMODEL + h * DKH];
      v8f s0 = {}, s1 = {};
#pragma unroll
      for (int c = 0; c < 2; ++c) {
        v16bf kf0 = load_frag_bf16(kr0 + c * 32, o);
        v16bf kf1 = load_frag_bf16(kr1 + c * 32, o);
        s0 = wmma_bf16(qf[hh][c], kf0, s0);
        s1 = wmma_bf16(qf[hh][c], kf1, s1);
      }
#pragma unroll
      for (int v = 0; v < 8; ++v) {
        e[hh][0][v] = __expf(s0[v] * 0.125f);   // 1/sqrt(64)
        e[hh][1][v] = __expf(s1[v] * 0.125f);
      }
    }

    // ---- cross-head denominator: sum over all 16 heads (8 waves) ----
#pragma unroll
    for (int v = 0; v < 8; ++v) {
      const int q = qr + v;
      atomicAdd(&denom[q * 32 + l15],      e[0][0][v] + e[1][0][v]);
      atomicAdd(&denom[q * 32 + l15 + 16], e[0][1][v] + e[1][1][v]);
    }
    __syncthreads();

    // ---- normalize, stage probs in LDS (C-layout -> row-major [q][k]) ----
    float r0[8], r1[8];
#pragma unroll
    for (int v = 0; v < 8; ++v) {
      const int q = qr + v;
      r0[v] = __builtin_amdgcn_rcpf(denom[q * 32 + l15]);
      r1[v] = __builtin_amdgcn_rcpf(denom[q * 32 + l15 + 16]);
    }
#pragma unroll
    for (int hh = 0; hh < 2; ++hh) {
      const int h = h0 + hh;
#pragma unroll
      for (int v = 0; v < 8; ++v) {
        const int q = qr + v;
        attnLds[h * 512 + q * 32 + l15]      = (__bf16)(e[hh][0][v] * r0[v]);
        attnLds[h * 512 + q * 32 + l15 + 16] = (__bf16)(e[hh][1][v] * r1[v]);
      }
    }
    __syncthreads();

    // ---- attn @ V : A = probs (16q x 32k), B^T rows = Vt[b][e][k-range] ----
#pragma unroll
    for (int hh = 0; hh < 2; ++hh) {
      const int h = h0 + hh;
      v16bf af = load_frag_bf16(&attnLds[h * 512 + l15 * 32], o);
#pragma unroll
      for (int nt = 0; nt < 4; ++nt) {
        const bf16_t* vr =
            Vt + (size_t)(b * DMODEL + h * DKH + nt * 16 + l15) * SEQL + k0;
        v16bf vf = load_frag_bf16(vr, o);
        outacc[hh][nt] = wmma_bf16(af, vf, outacc[hh][nt]);
      }
    }
  }

  // ---- write per-head context back as f32 [SB, DMODEL] ----
#pragma unroll
  for (int hh = 0; hh < 2; ++hh)
#pragma unroll
    for (int nt = 0; nt < 4; ++nt)
#pragma unroll
      for (int v = 0; v < 8; ++v) {
        const int q   = qr + v;
        const int m   = (q0 + q) * BATCHN + b;
        const int col = (h0 + hh) * DKH + nt * 16 + l15;
        AttnOut[(size_t)m * DMODEL + col] = outacc[hh][nt][v];
      }
}

// ---------------------------------------------------------------------------
extern "C" void kernel_launch(void* const* d_in, const int* in_sizes, int n_in,
                              void* d_out, int out_size, void* d_ws, size_t ws_size,
                              hipStream_t stream) {
  const float* query  = (const float*)d_in[0];
  const float* key    = (const float*)d_in[1];
  const float* value  = (const float*)d_in[2];
  const float* Wsplit = (const float*)d_in[3];
  const float* bsplit = (const float*)d_in[4];
  const float* Wo     = (const float*)d_in[5];
  const float* bo     = (const float*)d_in[6];
  float* out = (float*)d_out;

  // Workspace: Qb/Kb bf16 row-major, Vt bf16 transposed, AttnOut f32 (~84 MB).
  char* ws = (char*)d_ws;
  bf16_t* Qb      = (bf16_t*)(ws);
  bf16_t* Kb      = (bf16_t*)(ws + (size_t)SB * DMODEL * 2);
  bf16_t* Vt      = (bf16_t*)(ws + (size_t)SB * DMODEL * 4);
  float*  AttnOut = (float*) (ws + (size_t)SB * DMODEL * 6);

  dim3 blk(256);
  dim3 gGemm(1024);   // 8192 32x32 wave-tiles / 8 waves per block
  dim3 gAttn(512);    // BATCH * SEQ/16

  gemm_kernel<<<gGemm, blk, 0, stream>>>(query, Wsplit, bsplit, Qb, nullptr, nullptr, 0);
  gemm_kernel<<<gGemm, blk, 0, stream>>>(key,   Wsplit, bsplit, Kb, nullptr, nullptr, 0);
  gemm_kernel<<<gGemm, blk, 0, stream>>>(value, Wsplit, bsplit, nullptr, Vt, nullptr, 1);
  attn_kernel<<<gAttn, blk, 0, stream>>>(Qb, Kb, Vt, AttnOut);
  gemm_kernel<<<gGemm, blk, 0, stream>>>(AttnOut, Wo, bo, nullptr, nullptr, out, 2);
}
